// Megnet_NodeModel_42606075576611
// MI455X (gfx1250) — matrix operand order repaired and checked by reading, not compiled, not run on hardware
//
#include <hip/hip_runtime.h>

typedef __attribute__((ext_vector_type(16))) _Float16 v16h;
typedef __attribute__((ext_vector_type(8)))  _Float16 h8;
typedef __attribute__((ext_vector_type(8)))  float    v8f;

#define DFEAT 64          // node/edge/global feature dim
#define K1    192         // 3*D, GEMM1 reduction dim
#define W1S   (K1 + 16)   // padded k-stride (halves) for transposed W1 (416B rows, 16B aligned)
#define W2S   (DFEAT + 16)// padded k-stride (halves) for transposed W2 (160B rows, 16B aligned)
#define HSTR  72          // padded row stride (halves) for h tile (144B rows, 16B aligned)

// ---------------------------------------------------------------------------
// Kernel 1: zero the scatter accumulators (sums[N*64] ++ cnt[N]) in d_ws
// ---------------------------------------------------------------------------
__global__ void zero_f32_kernel(float* __restrict__ p, long long n) {
  long long i = (long long)blockIdx.x * blockDim.x + threadIdx.x;
  if (i < n) p[i] = 0.0f;
}

// ---------------------------------------------------------------------------
// Kernel 2: scatter-add edge_attr rows by source node + per-node edge count.
// One thread per (edge, feature): coalesced edge_attr reads; hardware
// global_atomic_add_f32 (unsafeAtomicAdd) into L2-resident accumulators
// (sums is 12.8 MB << 192 MB L2). This kernel is the bandwidth roofline:
// ~205 MB edge reads => ~10 us at 23.3 TB/s.
// ---------------------------------------------------------------------------
__global__ void scatter_add_kernel(const long long* __restrict__ edge_src,
                                   const float* __restrict__ edge_attr,
                                   float* __restrict__ sums,
                                   float* __restrict__ cnt,
                                   long long n_edges) {
  long long gid = (long long)blockIdx.x * blockDim.x + threadIdx.x;
  if (gid >= n_edges * DFEAT) return;
  int e = (int)(gid >> 6);
  int f = (int)(gid & (DFEAT - 1));
  int s = (int)edge_src[e];                 // edge_index row 0 (src), int64
  unsafeAtomicAdd(&sums[(long long)s * DFEAT + f], edge_attr[gid]);
  if (f == 0) unsafeAtomicAdd(&cnt[s], 1.0f);
}

// ---------------------------------------------------------------------------
// Kernel 3: fused  relu(relu([x | sums/cnt | u[batch]] @ W1 + b1) @ W2 + b2)
// 256 threads = 8 waves; each wave owns one 16-row M-tile.
// W1/W2 staged TRANSPOSED ([col][k]) in LDS as f16 so every WMMA B-operand
// fragment is two contiguous ds_load_b128 per lane. A operands come from
// vectorized float4 global loads (GEMM1) / contiguous ds_load_b128 (GEMM2).
// ---------------------------------------------------------------------------
__global__ __launch_bounds__(256) void megnet_mlp_wmma(
    const float* __restrict__ x,
    const float* __restrict__ sums,
    const float* __restrict__ cnt,
    const float* __restrict__ u,
    const long long* __restrict__ batch,
    const float* __restrict__ W1, const float* __restrict__ b1,
    const float* __restrict__ W2, const float* __restrict__ b2,
    float* __restrict__ out, int n_nodes)
{
  __shared__ __align__(16) _Float16 w1t[DFEAT][W1S];      // ~26 KB, [col][k]
  __shared__ __align__(16) _Float16 w2t[DFEAT][W2S];      // ~10 KB, [col][k]
  __shared__ __align__(16) _Float16 hsh[8][16][HSTR];     // ~18 KB, [wave][m][k]

  const int tid  = threadIdx.x;
  const int lane = tid & 31;
  const int wave = tid >> 5;
  const int l15  = lane & 15;
  const int hi   = lane >> 4;                   // 0: lanes 0-15, 1: lanes 16-31

  // Stage W1/W2 transposed into LDS as f16. Global reads coalesced
  // (consecutive tid -> consecutive col of one k-row).
  for (int i = tid; i < K1 * DFEAT; i += 256) {
    int k = i >> 6, col = i & 63;
    w1t[col][k] = (_Float16)W1[i];
  }
  for (int i = tid; i < DFEAT * DFEAT; i += 256) {
    int k = i >> 6, col = i & 63;
    w2t[col][k] = (_Float16)W2[i];
  }
  __syncthreads();

  // This wave's 16-row tile; clamp tail rows (keeps EXEC full for WMMA),
  // predicate only the final stores.
  const long long row_base = ((long long)blockIdx.x * 8 + wave) * 16;
  long long r = row_base + l15;
  if (r >= n_nodes) r = n_nodes - 1;

  const float* xrow = x    + r * DFEAT;
  const float* srow = sums + r * DFEAT;
  // scatter-mean scale: hardware v_rcp_f32 instead of IEEE divide sequence;
  // zero-count rows stay zero (sums row is zero).
  const float  inv  = __builtin_amdgcn_rcpf(fmaxf(cnt[r], 1.0f));
  const float* urow = u + (long long)batch[r] * DFEAT;

  // Load 8 consecutive elements of the virtual [x | v_e | u[batch]] row
  // starting at k (k is 8-aligned, run never crosses a 64-segment boundary,
  // so the segment choice is wave-uniform per k-step).
  auto load8 = [&](int k) -> h8 {
    const float* p; float sc;
    if (k < DFEAT)          { p = xrow + k;             sc = 1.0f; }
    else if (k < 2 * DFEAT) { p = srow + (k - DFEAT);   sc = inv;  }
    else                    { p = urow + (k - 2*DFEAT); sc = 1.0f; }
    float4 lo = *(const float4*)p;
    float4 hv = *(const float4*)(p + 4);
    h8 rv;
    rv[0] = (_Float16)(lo.x * sc); rv[1] = (_Float16)(lo.y * sc);
    rv[2] = (_Float16)(lo.z * sc); rv[3] = (_Float16)(lo.w * sc);
    rv[4] = (_Float16)(hv.x * sc); rv[5] = (_Float16)(hv.y * sc);
    rv[6] = (_Float16)(hv.z * sc); rv[7] = (_Float16)(hv.w * sc);
    return rv;
  };

  auto cat16 = [](h8 a, h8 b) -> v16h {
    return __builtin_shufflevector(a, b, 0, 1, 2, 3, 4, 5, 6, 7,
                                         8, 9, 10, 11, 12, 13, 14, 15);
  };

  // ---------------- GEMM1: comb @ W1 ----------------
  v8f acc[4] = {v8f{}, v8f{}, v8f{}, v8f{}};
  #pragma unroll
  for (int ks = 0; ks < K1 / 32; ++ks) {
    // A tile 16x32 f16: lane(l15,hi) holds row l15, K = ks*32 + hi*16 + j
    const int ka = ks * 32 + hi * 16;
    v16h a = cat16(load8(ka), load8(ka + 8));

    #pragma unroll
    for (int nt = 0; nt < 4; ++nt) {
      // B tile 32x16 f16: lane holds col nt*16+l15, K = ks*32 + hi*16 + j
      const int col = nt * 16 + l15;
      const h8* bp  = (const h8*)&w1t[col][ks * 32 + hi * 16];  // 16B aligned
      v16h b = cat16(bp[0], bp[1]);                              // 2x ds_load_b128
      acc[nt] = __builtin_amdgcn_wmma_f32_16x16x32_f16(
          false, a, false, b, (short)0, acc[nt], false, false);
    }
  }

  // bias + relu, spill h to LDS (f16, row-major) to re-layout C -> A.
  // C layout: vgpr j, lanes 0-15 -> (M=j, N=l15+nt*16); lanes 16-31 -> (M=j+8)
  #pragma unroll
  for (int nt = 0; nt < 4; ++nt) {
    const int col  = nt * 16 + l15;
    const float bb = b1[col];
    #pragma unroll
    for (int j = 0; j < 8; ++j) {
      float v = fmaxf(acc[nt][j] + bb, 0.0f);
      hsh[wave][hi * 8 + j][col] = (_Float16)v;
    }
  }
  __syncthreads();

  // ---------------- GEMM2: h @ W2 ----------------
  const _Float16* hrow = &hsh[wave][l15][0];   // row base 144B => 16B aligned
  v8f acc2[4] = {v8f{}, v8f{}, v8f{}, v8f{}};
  #pragma unroll
  for (int ks = 0; ks < DFEAT / 32; ++ks) {
    const int ka = ks * 32 + hi * 16;
    const h8* a0 = (const h8*)(hrow + ka);
    const h8* a1 = (const h8*)(hrow + ka + 8);
    v16h a = cat16(*a0, *a1);                                    // 2x ds_load_b128

    #pragma unroll
    for (int nt = 0; nt < 4; ++nt) {
      const int col = nt * 16 + l15;
      const h8* bp  = (const h8*)&w2t[col][ks * 32 + hi * 16];
      v16h b = cat16(bp[0], bp[1]);
      acc2[nt] = __builtin_amdgcn_wmma_f32_16x16x32_f16(
          false, a, false, b, (short)0, acc2[nt], false, false);
    }
  }

  // bias + relu + store (16 consecutive floats per half-wave per vgpr row)
  #pragma unroll
  for (int nt = 0; nt < 4; ++nt) {
    const int col  = nt * 16 + l15;
    const float bb = b2[col];
    #pragma unroll
    for (int j = 0; j < 8; ++j) {
      const long long orow = row_base + hi * 8 + j;
      if (orow < n_nodes)
        out[orow * DFEAT + col] = fmaxf(acc2[nt][j] + bb, 0.0f);
    }
  }
}

// ---------------------------------------------------------------------------
extern "C" void kernel_launch(void* const* d_in, const int* in_sizes, int n_in,
                              void* d_out, int out_size, void* d_ws, size_t ws_size,
                              hipStream_t stream) {
  const float*     x          = (const float*)d_in[0];
  const long long* edge_index = (const long long*)d_in[1];   // [2, E] int64
  const float*     edge_attr  = (const float*)d_in[2];
  const float*     u          = (const float*)d_in[3];
  const long long* batch      = (const long long*)d_in[4];
  const float*     W1         = (const float*)d_in[5];
  const float*     b1         = (const float*)d_in[6];
  const float*     W2         = (const float*)d_in[7];
  const float*     b2         = (const float*)d_in[8];
  float*           out        = (float*)d_out;

  const int n_nodes = in_sizes[0] / DFEAT;
  const long long n_edges = in_sizes[2] / DFEAT;

  float* sums = (float*)d_ws;                       // [N, 64]
  float* cnt  = sums + (long long)n_nodes * DFEAT;  // [N]

  // 1) zero accumulators
  const long long nz = (long long)n_nodes * DFEAT + n_nodes;
  zero_f32_kernel<<<(unsigned)((nz + 255) / 256), 256, 0, stream>>>(sums, nz);

  // 2) scatter-add over source nodes (edge_index row 0 is src)
  const long long work = n_edges * DFEAT;
  scatter_add_kernel<<<(unsigned)((work + 255) / 256), 256, 0, stream>>>(
      edge_index, edge_attr, sums, cnt, n_edges);

  // 3) fused WMMA MLP: 8 x 16-row tiles per 256-thread block
  const int ntiles  = (n_nodes + 15) / 16;
  const int nblocks = (ntiles + 7) / 8;
  megnet_mlp_wmma<<<nblocks, 256, 0, stream>>>(
      x, sums, cnt, u, batch, W1, b1, W2, b2, out, n_nodes);
}
